// QuantumCoherenceLayer_11579231830139
// MI455X (gfx1250) — matrix operand order repaired and checked by reading, compiled, not run on hardware
//
#include <hip/hip_runtime.h>

#define EPSF 1e-8f
#define DDIM 1024
#define ROWS_PER_WG 16
#define NWAVES 16
#define NTHREADS 512
#define A_STRIDE 1032   // 1024 + 8 bf16 pad (16B) to spread LDS banks

typedef __attribute__((ext_vector_type(16))) __bf16 bf16x16;
typedef __attribute__((ext_vector_type(8)))  float  floatx8;

union Frag32 {            // 32-byte WMMA 16-bit operand (16 bf16 per lane)
    uint4   q[2];
    bf16x16 v;
};

__device__ __forceinline__ unsigned short f2bf(float f) {
    unsigned int u = __float_as_uint(f);
    u = (u + 0x7FFFu + ((u >> 16) & 1u)) >> 16;   // round to nearest even
    return (unsigned short)u;
}

// ---------------------------------------------------------------------------
// Prep: phase_matrix (f32, [k][n] row-major) -> frag-blocked bf16 "Bt" in ws.
// Block (ntile, kstep) = 1024 contiguous bytes; within block lane L owns 32B:
//   n = ntile*16 + (L&15);  K = kstep*32 + (L<16 ? 0 : 16) + j, j=0..15
// which matches the contiguous-K A-fragment order used in qc_main.
// ---------------------------------------------------------------------------
__global__ void qc_prep_B(const float* __restrict__ P,
                          unsigned short* __restrict__ Bt) {
    __shared__ float tile[64][65];
    const int tid = threadIdx.x;                  // 256 threads
    const int tk  = blockIdx.x & 15;              // 64-wide k tile
    const int tn  = blockIdx.x >> 4;              // 64-wide n tile
    const int k0  = tk * 64, n0 = tn * 64;

#pragma unroll
    for (int i = 0; i < 4; ++i) {                 // coalesced f32x4 reads
        int f4 = tid + i * 256;                   // 0..1023
        int r  = f4 >> 4;
        int c4 = f4 & 15;
        float4 v = *(const float4*)&P[(size_t)(k0 + r) * DDIM + n0 + c4 * 4];
        tile[r][c4 * 4 + 0] = v.x;
        tile[r][c4 * 4 + 1] = v.y;
        tile[r][c4 * 4 + 2] = v.z;
        tile[r][c4 * 4 + 3] = v.w;
    }
    __syncthreads();

    // 256 writer slots: (ktile 2) x (nt 4) x (lane 32); 32B per slot, coalesced
    const int lane  = tid & 31;
    const int nt    = (tid >> 5) & 3;
    const int ktile = tid >> 7;                   // 0..1
    const int n_loc = nt * 16 + (lane & 15);
    const int kb    = ktile * 32 + ((lane & 16) ? 16 : 0);

    unsigned short h[16];
#pragma unroll
    for (int j = 0; j < 16; ++j) h[j] = f2bf(tile[kb + j][n_loc]);

    uint4 q0, q1;
    q0.x = h[0]  | (h[1]  << 16); q0.y = h[2]  | (h[3]  << 16);
    q0.z = h[4]  | (h[5]  << 16); q0.w = h[6]  | (h[7]  << 16);
    q1.x = h[8]  | (h[9]  << 16); q1.y = h[10] | (h[11] << 16);
    q1.z = h[12] | (h[13] << 16); q1.w = h[14] | (h[15] << 16);

    const int ntile_g = tn * 4 + nt;              // 0..63
    const int ks_g    = tk * 2 + ktile;           // 0..31
    uint4* dst = (uint4*)(Bt + (size_t)(ntile_g * 32 + ks_g) * 512 + lane * 16);
    dst[0] = q0;
    dst[1] = q1;
}

// ---------------------------------------------------------------------------
// Fragment load helpers for the fused main kernel
// ---------------------------------------------------------------------------
__device__ __forceinline__ void load_a_frag(Frag32& a,
                                            const unsigned short* Abf,
                                            int m, int kOffA, int ks) {
    const unsigned short* ap = &Abf[m * A_STRIDE + ks * 32 + kOffA];
    a.q[0] = *(const uint4*)(ap);                 // K +0..7
    a.q[1] = *(const uint4*)(ap + 8);             // K +8..15
}

__device__ __forceinline__ void load_b_frags(Frag32* b,
                                             const unsigned short* __restrict__ Bt,
                                             int wave, int lane, int ks, int ksPre) {
#pragma unroll
    for (int nt = 0; nt < 4; ++nt) {
        const int ntile = wave * 4 + nt;
        const uint4* blk = (const uint4*)(Bt + (size_t)(ntile * 32 + ks) * 512);
        __builtin_prefetch(Bt + (size_t)(ntile * 32 + ksPre) * 512 + lane * 16, 0, 3);
        b[nt].q[0] = blk[lane * 2];
        b[nt].q[1] = blk[lane * 2 + 1];
    }
}

// ---------------------------------------------------------------------------
// Fused main: coherence + WMMA GEMM + rotation/normalize + 2x layernorm.
// One workgroup = 16 rows x full 1024 cols. 16 waves; wave w owns cols
// [w*64, w*64+64) (4 WMMA N-tiles).
// ---------------------------------------------------------------------------
__global__ __launch_bounds__(NTHREADS)
void qc_main(const float* __restrict__ x_real,
             const float* __restrict__ x_imag,
             const unsigned short* __restrict__ Bt,
             const float* __restrict__ cfp,
             const float* __restrict__ ln_w,
             const float* __restrict__ ln_b,
             float* __restrict__ out_r,
             float* __restrict__ out_i) {
    // 128KB buffer: phase 1-2 = bf16 A tile [16][A_STRIDE]; phase 3 = f32 stage
    __shared__ __align__(16) unsigned char smem[131072];
    __shared__ float s_sc[ROWS_PER_WG], s_ss[ROWS_PER_WG], s_scale[ROWS_PER_WG];

    unsigned short* Abf   = (unsigned short*)smem;
    float*          stg_r = (float*)smem;
    float*          stg_i = stg_r + ROWS_PER_WG * DDIM;

    const int tid  = threadIdx.x;
    const int wave = tid >> 5;
    const int lane = tid & 31;
    const int row0 = blockIdx.x * ROWS_PER_WG;

    if (tid < ROWS_PER_WG) { s_sc[tid] = 0.0f; s_ss[tid] = 0.0f; }
    __syncthreads();

    // -------- Phase 1: load x rows, bf16-convert A, coherence partials ------
#pragma unroll
    for (int i = 0; i < 8; ++i) {
        int f4 = tid + i * NTHREADS;              // 0..4095 (float4 index)
        int r  = f4 >> 8;                         // row in tile
        int c4 = f4 & 255;
        size_t g = (size_t)(row0 + r) * DDIM + c4 * 4;
        float4 xr = *(const float4*)&x_real[g];
        float4 xi = *(const float4*)&x_imag[g];

        unsigned long long pk =
            (unsigned long long)f2bf(xr.x)
          | ((unsigned long long)f2bf(xr.y) << 16)
          | ((unsigned long long)f2bf(xr.z) << 32)
          | ((unsigned long long)f2bf(xr.w) << 48);
        *(unsigned long long*)&Abf[r * A_STRIDE + c4 * 4] = pk;

        float psc = 0.0f, pss = 0.0f;
        float ar, ai, inv;
        ar = xr.x + EPSF; ai = xi.x + EPSF; inv = rsqrtf(ar*ar + ai*ai); psc += ar*inv; pss += ai*inv;
        ar = xr.y + EPSF; ai = xi.y + EPSF; inv = rsqrtf(ar*ar + ai*ai); psc += ar*inv; pss += ai*inv;
        ar = xr.z + EPSF; ai = xi.z + EPSF; inv = rsqrtf(ar*ar + ai*ai); psc += ar*inv; pss += ai*inv;
        ar = xr.w + EPSF; ai = xi.w + EPSF; inv = rsqrtf(ar*ar + ai*ai); psc += ar*inv; pss += ai*inv;
        atomicAdd(&s_sc[r], psc);
        atomicAdd(&s_ss[r], pss);
    }
    __syncthreads();

    if (tid < ROWS_PER_WG) {
        float sc = s_sc[tid], ss = s_ss[tid];
        float coh = (sc * sc + ss * ss) / ((float)DDIM * (float)DDIM);
        s_scale[tid] = cfp[0] * (1.0f - coh);
    }
    __syncthreads();

    // -------- Phase 2: WMMA GEMM (pa = x_real @ P), software-pipelined ------
    const int m      = lane & 15;
    const int kOffA  = (lane & 16) ? 16 : 0;      // A layout: contiguous 16 K per half
    floatx8 acc[4] = {};

    Frag32 a0, a1, b0[4], b1[4];
    load_a_frag(a0, Abf, m, kOffA, 0);
    load_b_frags(b0, Bt, wave, lane, 0, 1);

    for (int ks = 0; ks < 32; ks += 2) {
        // preload ks+1 into buffer 1 while buffer 0 feeds the WMMAs
        load_a_frag(a1, Abf, m, kOffA, ks + 1);
        load_b_frags(b1, Bt, wave, lane, ks + 1, (ks + 2 < 32) ? ks + 2 : 31);
#pragma unroll
        for (int nt = 0; nt < 4; ++nt)
            acc[nt] = __builtin_amdgcn_wmma_f32_16x16x32_bf16(
                false, a0.v, false, b0[nt].v, (short)0, acc[nt], false, false);

        // preload ks+2 into buffer 0 (clamped, branch-free on last iteration)
        const int k2 = (ks + 2 < 32) ? ks + 2 : 31;
        load_a_frag(a0, Abf, m, kOffA, k2);
        load_b_frags(b0, Bt, wave, lane, k2, (k2 + 1 < 32) ? k2 + 1 : 31);
#pragma unroll
        for (int nt = 0; nt < 4; ++nt)
            acc[nt] = __builtin_amdgcn_wmma_f32_16x16x32_bf16(
                false, a1.v, false, b1[nt].v, (short)0, acc[nt], false, false);
    }
    __syncthreads();   // all waves done reading Abf before stage reuse

    // -------- Phase 3: rotate, unit-modulus, stage to LDS -------------------
#pragma unroll
    for (int nt = 0; nt < 4; ++nt) {
        const int n = wave * 64 + nt * 16 + m;
#pragma unroll
        for (int v = 0; v < 8; ++v) {
            const int r = v + ((lane & 16) ? 8 : 0);   // C layout: M = v / v+8
            float pa = acc[nt][v] * s_scale[r];
            float s, c;
            __sincosf(pa, &s, &c);
            size_t g = (size_t)(row0 + r) * DDIM + n;
            float xr = x_real[g];
            float xi = x_imag[g];
            float nr = xr * c - xi * s;
            float ni = xr * s + xi * c;
            float inv = rsqrtf(nr * nr + ni * ni + EPSF);
            stg_r[r * DDIM + n] = nr * inv;
            stg_i[r * DDIM + n] = ni * inv;
        }
    }
    __syncthreads();

    // -------- Phase 4: per-row layernorm (wave w handles row w) -------------
    {
        const int r = wave;
        float sr = 0.0f, sr2 = 0.0f, si = 0.0f, si2 = 0.0f;
#pragma unroll
        for (int i = 0; i < DDIM / 32; ++i) {
            float vr = stg_r[r * DDIM + i * 32 + lane];
            float vi = stg_i[r * DDIM + i * 32 + lane];
            sr += vr; sr2 += vr * vr;
            si += vi; si2 += vi * vi;
        }
#pragma unroll
        for (int off = 16; off >= 1; off >>= 1) {
            sr  += __shfl_xor(sr,  off, 32);
            sr2 += __shfl_xor(sr2, off, 32);
            si  += __shfl_xor(si,  off, 32);
            si2 += __shfl_xor(si2, off, 32);
        }
        const float invD  = 1.0f / (float)DDIM;
        const float mu_r  = sr * invD;
        const float mu_i  = si * invD;
        const float is_r  = rsqrtf(sr2 * invD - mu_r * mu_r + EPSF);
        const float is_i  = rsqrtf(si2 * invD - mu_i * mu_i + EPSF);
#pragma unroll
        for (int i = 0; i < DDIM / 32; ++i) {
            int col = i * 32 + lane;
            float w = ln_w[col];
            float b = ln_b[col];
            size_t g = (size_t)(row0 + r) * DDIM + col;
            out_r[g] = (stg_r[r * DDIM + col] - mu_r) * is_r * w + b;
            out_i[g] = (stg_i[r * DDIM + col] - mu_i) * is_i * w + b;
        }
    }
}

extern "C" void kernel_launch(void* const* d_in, const int* in_sizes, int n_in,
                              void* d_out, int out_size, void* d_ws, size_t ws_size,
                              hipStream_t stream) {
    const float* x_real = (const float*)d_in[0];
    const float* x_imag = (const float*)d_in[1];
    const float* P      = (const float*)d_in[2];
    const float* cf     = (const float*)d_in[3];
    const float* ln_w   = (const float*)d_in[4];
    const float* ln_b   = (const float*)d_in[5];

    const int M = in_sizes[0] / DDIM;             // 16384 rows
    float* out_r = (float*)d_out;
    float* out_i = out_r + (size_t)M * DDIM;

    unsigned short* Bt = (unsigned short*)d_ws;   // 2MB frag-blocked bf16 B

    qc_prep_B<<<dim3(256), dim3(256), 0, stream>>>(P, Bt);
    qc_main<<<dim3(M / ROWS_PER_WG), dim3(NTHREADS), 0, stream>>>(
        x_real, x_imag, Bt, cf, ln_w, ln_b, out_r, out_i);
}
